// FrequencyAttention_73624329388173
// MI455X (gfx1250) — matrix operand order repaired and checked by reading, not compiled, not run on hardware
//
#include <hip/hip_runtime.h>
#include <cstdint>

typedef __bf16 bf16;
typedef __attribute__((ext_vector_type(4)))  bf16  v4bf;
typedef __attribute__((ext_vector_type(8)))  bf16  v8bf;
typedef __attribute__((ext_vector_type(16))) bf16  v16bf;
typedef __attribute__((ext_vector_type(8)))  float v8f;
typedef __attribute__((ext_vector_type(4)))  unsigned int u32x4;
typedef __attribute__((ext_vector_type(8)))  unsigned int u32x8;

#define S_LEN   1024
#define D_MODEL 1024
#define NB      4
#define NH      16
#define NM      64
#define HDIM    64

__device__ __forceinline__ v8f vzero8() {
  v8f z;
#pragma unroll
  for (int i = 0; i < 8; ++i) z[i] = 0.f;
  return z;
}

// ---- WMMA fragment loaders (bf16, 16x16x32) -------------------------------
// A (16x32 MxK): lane<16 -> M=lane, K=0..7 & 16..23; lane>=16 -> K=8..15 & 24..31.
__device__ __forceinline__ v16bf frag_a(const bf16* tile, int ld, int mBase,
                                        int kOff, int lane) {
  const bf16* p = tile + (mBase + (lane & 15)) * ld + kOff + ((lane >> 4) << 3);
  v8bf lo = *(const v8bf*)p;
  v8bf hi = *(const v8bf*)(p + 16);
  v16bf r;
#pragma unroll
  for (int i = 0; i < 8; ++i) { r[i] = lo[i]; r[i + 8] = hi[i]; }
  return r;
}
// B (32x16 KxN), LDS stored [n][k]: lane<16 -> N=lane, K=0..7 & 16..23; lane>=16 -> K=8..15 & 24..31.
__device__ __forceinline__ v16bf frag_b(const bf16* tile, int ld, int nBase,
                                        int kOff, int lane) {
  const bf16* p = tile + (nBase + (lane & 15)) * ld + kOff + ((lane >> 4) << 3);
  v8bf lo = *(const v8bf*)p;
  v8bf hi = *(const v8bf*)(p + 16);
  v16bf r;
#pragma unroll
  for (int i = 0; i < 8; ++i) { r[i] = lo[i]; r[i + 8] = hi[i]; }
  return r;
}
__device__ __forceinline__ v8f wmma_bf16(v16bf a, v16bf b, v8f c) {
  return __builtin_amdgcn_wmma_f32_16x16x32_bf16(false, a, false, b, (short)0, c,
                                                 false, false);
}

// ---- Tensor Data Mover: async copy of a [rows x tile_w] bf16 tile to LDS --
// D# per CDNA5 ISA. data_size=2B. Padding adds 16B after
// every (1<<pad_ivl) 8-byte units, producing the padded LDS row stride that the
// WMMA fragment loaders expect (tile_w=32 -> stride 40; tile_w=128 -> 136).
__device__ __forceinline__ void tdm_load_tile(unsigned int lds_off,
                                              const bf16* gptr,
                                              unsigned int tile_w_elems,
                                              unsigned int rows,
                                              unsigned int row_stride_elems,
                                              unsigned int pad_ivl) {
  unsigned long long ga = (unsigned long long)(uintptr_t)gptr;
  u32x4 g0;
  g0[0] = 1u;                                        // count=1 (valid descriptor)
  g0[1] = lds_off;                                   // lds_addr (bytes)
  g0[2] = (unsigned int)ga;                          // global_addr[31:0]
  g0[3] = (unsigned int)((ga >> 32) & 0x01FFFFFFu)   // global_addr[56:32]
          | 0x80000000u;                             // type=2 ("image")
  u32x8 g1;
  g1[0] = (1u << 16)            // data_size = 2 bytes
        | (1u << 20)            // pad_enable
        | (pad_ivl << 22)       // pad_interval
        | (3u << 25);           // pad_amount: 4 DWORDs (16B)
  g1[1] = (row_stride_elems & 0xFFFFu) << 16;        // tensor_dim0[15:0]
  g1[2] = (row_stride_elems >> 16) | 0xFFFF0000u;    // tensor_dim0 hi | dim1 lo
  g1[3] = 0x7FFFu | (tile_w_elems << 16);            // dim1 hi | tile_dim0
  g1[4] = rows;                                      // tile_dim1 | tile_dim2=0
  g1[5] = row_stride_elems;                          // tensor_dim0_stride lo32
  g1[6] = 0u;                                        // stride hi | dim1_stride lo
  g1[7] = 0u;
  asm volatile("tensor_load_to_lds %0, %1" : : "s"(g0), "s"(g1) : "memory");
}

// ---- bf16 GEMM: C[m,n] = sum_k A[m,k]*W[n,k] + bias[n] --------------------
// A:[M,K] bf16, W:[N,K] bf16 row-major. 128x128 block, 8 wave32s, TDM-fed
// double-buffered LDS (tiles never touch VGPRs on the way in).
template <bool OUT_BF16>
__global__ __launch_bounds__(256) void wmma_gemm_tdm(
    const bf16* __restrict__ A, const bf16* __restrict__ W,
    const float* __restrict__ bias, void* __restrict__ Cp, int M, int N, int K) {
  __shared__ bf16 aT[2][128 * 40];
  __shared__ bf16 bT[2][128 * 40];
  const int tid = threadIdx.x, lane = tid & 31, wv = tid >> 5;
  const int tM = blockIdx.y * 128, tN = blockIdx.x * 128;
  const int wm = (wv >> 1) * 32, wn = (wv & 1) * 64;
  v8f acc[2][4];
#pragma unroll
  for (int i = 0; i < 2; ++i)
#pragma unroll
    for (int j = 0; j < 4; ++j) acc[i][j] = vzero8();

  if (wv == 0) {
    tdm_load_tile((unsigned int)(uintptr_t)&aT[0][0], A + (size_t)tM * K, 32, 128, K, 3);
    tdm_load_tile((unsigned int)(uintptr_t)&bT[0][0], W + (size_t)tN * K, 32, 128, K, 3);
    __builtin_amdgcn_s_wait_tensorcnt(0);
  }
  __syncthreads();

  int cur = 0;
  for (int kt = 0; kt < K; kt += 32) {
    if (wv == 0 && kt + 32 < K) {  // prefetch next K-slab via TDM (async)
      tdm_load_tile((unsigned int)(uintptr_t)&aT[cur ^ 1][0],
                    A + (size_t)tM * K + kt + 32, 32, 128, K, 3);
      tdm_load_tile((unsigned int)(uintptr_t)&bT[cur ^ 1][0],
                    W + (size_t)tN * K + kt + 32, 32, 128, K, 3);
    }
    v16bf af[2], bfr[4];
#pragma unroll
    for (int i = 0; i < 2; ++i) af[i] = frag_a(&aT[cur][0], 40, wm + i * 16, 0, lane);
#pragma unroll
    for (int j = 0; j < 4; ++j) bfr[j] = frag_b(&bT[cur][0], 40, wn + j * 16, 0, lane);
#pragma unroll
    for (int i = 0; i < 2; ++i)
#pragma unroll
      for (int j = 0; j < 4; ++j) acc[i][j] = wmma_bf16(af[i], bfr[j], acc[i][j]);
    if (wv == 0) __builtin_amdgcn_s_wait_tensorcnt(0);
    __syncthreads();
    cur ^= 1;
  }

#pragma unroll
  for (int i = 0; i < 2; ++i) {
    int mB = tM + wm + i * 16 + ((lane >> 4) << 3);
#pragma unroll
    for (int j = 0; j < 4; ++j) {
      int nC = tN + wn + j * 16 + (lane & 15);
      float bv = bias[nC];
#pragma unroll
      for (int r = 0; r < 8; ++r) {
        float v = acc[i][j][r] + bv;
        if (OUT_BF16)
          ((bf16*)Cp)[(size_t)(mB + r) * N + nC] = (bf16)v;
        else
          ((float*)Cp)[(size_t)(mB + r) * N + nC] = v;
      }
    }
  }
}

// ---- fp32 -> bf16 pre-conversion (query / w_qkv / w_out) ------------------
__global__ void f32_to_bf16(const float* __restrict__ src, bf16* __restrict__ dst,
                            int n) {
  int i = (blockIdx.x * blockDim.x + threadIdx.x) * 4;
  if (i >= n) return;
  float4 d = *(const float4*)(src + i);
  v4bf o;
  o[0] = (bf16)d.x; o[1] = (bf16)d.y; o[2] = (bf16)d.z; o[3] = (bf16)d.w;
  *(v4bf*)(dst + i) = o;
}

// ---- cos/sin basis (bf16) for the 64 retained frequencies -----------------
__global__ void gen_basis(bf16* __restrict__ cosF, bf16* __restrict__ sinF,
                          bf16* __restrict__ basisT) {
  int idx = blockIdx.x * blockDim.x + threadIdx.x;
  if (idx >= NM * S_LEN) return;
  int f = idx >> 10, s = idx & (S_LEN - 1);
  float th = 6.28318530717958647692f * (float)(f * s) / (float)S_LEN;
  float sn, cs;
  __sincosf(th, &sn, &cs);
  cosF[idx] = (bf16)cs;
  sinF[idx] = (bf16)sn;
  basisT[s * 128 + f] = (bf16)cs;
  basisT[s * 128 + 64 + f] = (bf16)sn;
}

// ---- forward DFT (64 freqs) + real filter, per (tensor, head, batch) ------
// A-tile (cos/sin basis slab) is TDM-fed; only the true transpose-scatter of
// the activation B-tile goes through VGPRs.
__global__ __launch_bounds__(256) void freq_transform(
    const bf16* __restrict__ qkv, const bf16* __restrict__ cosF,
    const bf16* __restrict__ sinF, const float* __restrict__ fq,
    const float* __restrict__ fk, const float* __restrict__ fv,
    float* __restrict__ comps) {
  __shared__ bf16 aT[128 * 40];
  __shared__ bf16 bT[64 * 40];
  const int t = blockIdx.x, h = blockIdx.y, b = blockIdx.z;
  const int tid = threadIdx.x, lane = tid & 31, wv = tid >> 5;
  const int wm = (wv >> 1) * 32, wn = (wv & 1) * 32;
  const bf16* X = qkv + (size_t)b * S_LEN * (3 * D_MODEL) + t * D_MODEL + h * HDIM;
  v8f acc[2][2];
#pragma unroll
  for (int i = 0; i < 2; ++i)
#pragma unroll
    for (int j = 0; j < 2; ++j) acc[i][j] = vzero8();

  for (int kt = 0; kt < S_LEN; kt += 32) {
    if (wv == 0) {  // A: cos rows -> LDS rows 0..63, sin rows -> 64..127
      tdm_load_tile((unsigned int)(uintptr_t)&aT[0], cosF + kt, 32, 64, S_LEN, 3);
      tdm_load_tile((unsigned int)(uintptr_t)&aT[64 * 40], sinF + kt, 32, 64, S_LEN, 3);
    }
    {  // B: data [s][hd] -> LDS [hd][s] (transpose scatter, bf16)
      int sl = tid >> 3, hg = (tid & 7) << 3;
      v8bf xv = *(const v8bf*)(X + (size_t)(kt + sl) * (3 * D_MODEL) + hg);
#pragma unroll
      for (int j = 0; j < 8; ++j) bT[(hg + j) * 40 + sl] = xv[j];
    }
    if (wv == 0) __builtin_amdgcn_s_wait_tensorcnt(0);
    __syncthreads();
    v16bf af[2], bfr[2];
#pragma unroll
    for (int i = 0; i < 2; ++i) af[i] = frag_a(aT, 40, wm + i * 16, 0, lane);
#pragma unroll
    for (int j = 0; j < 2; ++j) bfr[j] = frag_b(bT, 40, wn + j * 16, 0, lane);
#pragma unroll
    for (int i = 0; i < 2; ++i)
#pragma unroll
      for (int j = 0; j < 2; ++j) acc[i][j] = wmma_bf16(af[i], bfr[j], acc[i][j]);
    __syncthreads();
  }
  const float* wflt = (t == 0) ? fq : (t == 1) ? fk : fv;
#pragma unroll
  for (int i = 0; i < 2; ++i) {
    int fp = wm + i * 16 + ((lane >> 4) << 3);
#pragma unroll
    for (int j = 0; j < 2; ++j) {
      int hd = wn + j * 16 + (lane & 15);
#pragma unroll
      for (int r = 0; r < 8; ++r) {
        int fr = fp + r;
        int c = fr >> 6;  // 0 = Re, 1 = Im
        int f = fr & 63;
        float val = acc[i][j][r] * wflt[(h * NM + f) * HDIM + hd];
        comps[(size_t)((t * 2 + c) * NB * NH + b * NH + h) * (NM * HDIM) +
              f * HDIM + hd] = val;
      }
    }
  }
}

// ---- scores + softmax (includes the (S-M) zero-score padded entries) ------
__global__ __launch_bounds__(64) void scores_softmax(
    const float* __restrict__ comps, float* __restrict__ attn) {
  __shared__ float red[64];
  const int bh = blockIdx.x;
  const int f = threadIdx.x;
  const size_t P = (size_t)NB * NH * NM * HDIM;  // one component plane
  const size_t base = (size_t)bh * (NM * HDIM);
  const float* qr = comps + 0 * P + base;
  const float* qi = comps + 1 * P + base;
  const float* kr = comps + 2 * P + base;
  const float* ki = comps + 3 * P + base;
  float s = 0.f;
  for (int hd = 0; hd < HDIM; ++hd)
    s += qr[f * HDIM + hd] * kr[f * HDIM + hd] + qi[f * HDIM + hd] * ki[f * HDIM + hd];
  red[f] = s;
  __syncthreads();
  float mx = 0.f;  // zero-padded scores participate in the max
  for (int i = 0; i < 64; ++i) mx = fmaxf(mx, red[i]);
  float ex = __expf(s - mx);
  __syncthreads();
  red[f] = ex;
  __syncthreads();
  float sum = (float)(S_LEN - NM) * __expf(-mx);
  for (int i = 0; i < 64; ++i) sum += red[i];
  attn[bh * NM + f] = ex / sum;
}

// ---- inverse DFT with attn folded into V; bf16 output for final GEMM ------
// A-tile (basisT slab, 128x128) is TDM-fed with 16B pad per 256B row -> the
// 136-element LDS stride the fragment loaders use.
__global__ __launch_bounds__(256) void inverse_dft(
    const bf16* __restrict__ basisT, const float* __restrict__ comps,
    const float* __restrict__ attn, bf16* __restrict__ outT) {
  __shared__ bf16 aT[128 * 136];
  __shared__ bf16 bT[64 * 136];
  const int st = blockIdx.x, h = blockIdx.y, b = blockIdx.z;
  const int tid = threadIdx.x, lane = tid & 31, wv = tid >> 5;
  const int bh = b * NH + h;
  const size_t P = (size_t)NB * NH * NM * HDIM;
  const float* vr = comps + 4 * P + (size_t)bh * (NM * HDIM);
  const float* vi = comps + 5 * P + (size_t)bh * (NM * HDIM);
  const float* at = attn + bh * NM;
  if (wv == 0)
    tdm_load_tile((unsigned int)(uintptr_t)&aT[0],
                  basisT + (size_t)st * 128 * 128, 128, 128, 128, 5);
#pragma unroll
  for (int i = 0; i < 32; ++i) {  // B: AV[k][hd] -> LDS [hd][k], f32 math
    int idx = tid + (i << 8);
    int k = idx >> 6, hd = idx & 63;
    float av = (k < 64) ? at[k] * vr[k * HDIM + hd]
                        : at[k - 64] * vi[(k - 64) * HDIM + hd];
    bT[hd * 136 + k] = (bf16)av;
  }
  if (wv == 0) __builtin_amdgcn_s_wait_tensorcnt(0);
  __syncthreads();
  const int wm = (wv >> 1) * 32, wn = (wv & 1) * 32;
  v8f acc[2][2];
#pragma unroll
  for (int i = 0; i < 2; ++i)
#pragma unroll
    for (int j = 0; j < 2; ++j) acc[i][j] = vzero8();
#pragma unroll
  for (int ks = 0; ks < 128; ks += 32) {
    v16bf af[2], bfr[2];
#pragma unroll
    for (int i = 0; i < 2; ++i) af[i] = frag_a(aT, 136, wm + i * 16, ks, lane);
#pragma unroll
    for (int j = 0; j < 2; ++j) bfr[j] = frag_b(bT, 136, wn + j * 16, ks, lane);
#pragma unroll
    for (int i = 0; i < 2; ++i)
#pragma unroll
      for (int j = 0; j < 2; ++j) acc[i][j] = wmma_bf16(af[i], bfr[j], acc[i][j]);
  }
#pragma unroll
  for (int i = 0; i < 2; ++i) {
    int sg = st * 128 + wm + i * 16 + ((lane >> 4) << 3);
#pragma unroll
    for (int j = 0; j < 2; ++j) {
      int hd = wn + j * 16 + (lane & 15);
#pragma unroll
      for (int r = 0; r < 8; ++r)
        outT[((size_t)b * S_LEN + sg + r) * D_MODEL + h * HDIM + hd] =
            (bf16)acc[i][j][r];
    }
  }
}

extern "C" void kernel_launch(void* const* d_in, const int* in_sizes, int n_in,
                              void* d_out, int out_size, void* d_ws, size_t ws_size,
                              hipStream_t stream) {
  const float* query  = (const float*)d_in[0];
  const float* w_qkv  = (const float*)d_in[1];
  const float* b_qkv  = (const float*)d_in[2];
  const float* w_out  = (const float*)d_in[3];
  const float* b_out  = (const float*)d_in[4];
  const float* freq_q = (const float*)d_in[5];
  const float* freq_k = (const float*)d_in[6];
  const float* freq_v = (const float*)d_in[7];
  float* out = (float*)d_out;

  char* w = (char*)d_ws;
  auto alloc = [&](size_t bytes) {
    char* p = w;
    w += (bytes + 255) & ~(size_t)255;
    return p;
  };
  const size_t nQ = (size_t)NB * S_LEN * D_MODEL;       // 4.19M
  const size_t nWqkv = (size_t)3 * D_MODEL * D_MODEL;   // 3.15M
  const size_t nWout = (size_t)D_MODEL * D_MODEL;       // 1.05M
  bf16* queryH = (bf16*)alloc(nQ * 2);
  bf16* wqkvH  = (bf16*)alloc(nWqkv * 2);
  bf16* woutH  = (bf16*)alloc(nWout * 2);
  bf16* qkvH   = (bf16*)alloc((size_t)NB * S_LEN * 3 * D_MODEL * 2);
  bf16* cosF   = (bf16*)alloc((size_t)NM * S_LEN * 2);
  bf16* sinF   = (bf16*)alloc((size_t)NM * S_LEN * 2);
  bf16* basisT = (bf16*)alloc((size_t)S_LEN * 128 * 2);
  float* comps = (float*)alloc((size_t)6 * NB * NH * NM * HDIM * 4);
  float* attnW = (float*)alloc((size_t)NB * NH * NM * 4);
  bf16* outTH  = (bf16*)alloc(nQ * 2);

  f32_to_bf16<<<(int)(nQ / 4 + 255) / 256, 256, 0, stream>>>(query, queryH, (int)nQ);
  f32_to_bf16<<<(int)(nWqkv / 4 + 255) / 256, 256, 0, stream>>>(w_qkv, wqkvH, (int)nWqkv);
  f32_to_bf16<<<(int)(nWout / 4 + 255) / 256, 256, 0, stream>>>(w_out, woutH, (int)nWout);
  gen_basis<<<(NM * S_LEN + 255) / 256, 256, 0, stream>>>(cosF, sinF, basisT);

  wmma_gemm_tdm<true><<<dim3(3 * D_MODEL / 128, NB * S_LEN / 128), 256, 0, stream>>>(
      queryH, wqkvH, b_qkv, qkvH, NB * S_LEN, 3 * D_MODEL, D_MODEL);
  freq_transform<<<dim3(3, NH, NB), 256, 0, stream>>>(qkvH, cosF, sinF, freq_q,
                                                      freq_k, freq_v, comps);
  scores_softmax<<<NB * NH, 64, 0, stream>>>(comps, attnW);
  inverse_dft<<<dim3(S_LEN / 128, NH, NB), 256, 0, stream>>>(basisT, comps, attnW,
                                                             outTH);
  wmma_gemm_tdm<false><<<dim3(D_MODEL / 128, NB * S_LEN / 128), 256, 0, stream>>>(
      outTH, woutH, b_out, out, NB * S_LEN, D_MODEL, D_MODEL);
}
